// NachumModel_39359080301046
// MI455X (gfx1250) — matrix operand order repaired and checked by reading, not compiled, not run on hardware
//
#include <hip/hip_runtime.h>
#include <hip/hip_bf16.h>

// ---------------------------------------------------------------------------
// Problem constants (match the JAX reference)
// ---------------------------------------------------------------------------
#define NN        4096
#define X_DIM     256
#define Z_DIM     128
#define HDIM      512
#define NUM_ACT   4
#define KG_PAD    288          // 260 padded up to a multiple of 32
#define INV_T     10.0f        // 1 / TEMPERATURE
#define NSEG      4            // column segments for the fused softmax
#define SEG_COLS  (NN / NSEG)  // 1024 columns per segment
#define SEG_CHUNK (SEG_COLS / 16)

typedef _Float16 half8  __attribute__((ext_vector_type(8)));
typedef _Float16 half16 __attribute__((ext_vector_type(16)));
typedef float    float8 __attribute__((ext_vector_type(8)));

// ---------------------------------------------------------------------------
// WMMA fragment helpers (CDNA5 wave32 layouts, CDNA5 ISA §7.12.2)
// ---------------------------------------------------------------------------
__device__ __forceinline__ half16 frag_cat(half8 lo, half8 hi) {
  return __builtin_shufflevector(lo, hi, 0, 1, 2, 3, 4, 5, 6, 7,
                                         8, 9, 10, 11, 12, 13, 14, 15);
}

// A fragment: 16x32 f16 tile, row-major source, leading dim lda (elements).
// Lanes 0-15: row = lane, halves 0-15 = K 0..15
// Lanes 16-31: row = lane-16, halves 0-15 = K 16..31
__device__ __forceinline__ half16 load_a_frag(const _Float16* __restrict__ base,
                                              int lda, int lane) {
  int row = lane & 15;
  int kb  = (lane >> 4) << 4;          // 0 or 16
  const _Float16* p = base + (size_t)row * lda + kb;
  half8 lo = *reinterpret_cast<const half8*>(p);
  half8 hi = *reinterpret_cast<const half8*>(p + 8);
  return frag_cat(lo, hi);
}

// B fragment: 32x16 (KxN) where column n of B = row n of `base` (row-major,
// leading dim ldb). Lanes 0-15: col = lane, K = 0..15; lanes 16-31: K = 16..31.
__device__ __forceinline__ half16 load_b_frag(const _Float16* __restrict__ base,
                                              int ldb, int lane) {
  int col = lane & 15;
  int kb  = (lane >> 4) << 4;          // 0 or 16
  const _Float16* p = base + (size_t)col * ldb + kb;
  half8 lo = *reinterpret_cast<const half8*>(p);
  half8 hi = *reinterpret_cast<const half8*>(p + 8);
  return frag_cat(lo, hi);
}

__device__ __forceinline__ float8 wmma_f16(half16 a, half16 b, float8 c) {
  return __builtin_amdgcn_wmma_f32_16x16x32_f16(false, a, false, b,
                                                (short)0, c, false, false);
}

// ---------------------------------------------------------------------------
// CDNA5 async copy helpers (CDNA5 ISA)
// One 16x128 f16 tile (4KB) = 8 x b128 per lane, tracked by ASYNCcnt.
// ---------------------------------------------------------------------------
__device__ __forceinline__ void async_tile_load(const _Float16* gsrc,
                                                unsigned lds_base, int lane) {
  unsigned long long ga =
      (unsigned long long)(size_t)gsrc + (unsigned long long)(lane * 128);
  unsigned la = lds_base + (unsigned)(lane * 128);
#pragma unroll
  for (int i = 0; i < 8; ++i) {
    unsigned long long g = ga + (unsigned long long)(i * 16);
    unsigned l = la + (unsigned)(i * 16);
    asm volatile("global_load_async_to_lds_b128 %0, %1, off"
                 :: "v"(l), "v"(g)
                 : "memory");
  }
}
__device__ __forceinline__ void wait_async_le8() {
  asm volatile("s_wait_asynccnt 0x8" ::: "memory");
}
__device__ __forceinline__ void wait_async_0() {
  asm volatile("s_wait_asynccnt 0x0" ::: "memory");
}

// ---------------------------------------------------------------------------
// Prep kernels: f32 -> f16 conversion / concat+pad / weight transpose
// ---------------------------------------------------------------------------
__global__ void k_cvt_f16(const float* __restrict__ x, _Float16* __restrict__ y,
                          int n) {
  int i = blockIdx.x * blockDim.x + threadIdx.x;
  if (i < n) y[i] = (_Float16)x[i];
}

// xg[n][k]: k<256 -> sp, 256<=k<260 -> a1h, else 0
__global__ void k_build_xg(const float* __restrict__ sp,
                           const float* __restrict__ a1h,
                           _Float16* __restrict__ xg, int total) {
  int i = blockIdx.x * blockDim.x + threadIdx.x;
  if (i >= total) return;
  int n = i / KG_PAD, k = i % KG_PAD;
  float v = 0.0f;
  if (k < X_DIM)                 v = sp[(size_t)n * X_DIM + k];
  else if (k < X_DIM + NUM_ACT)  v = a1h[(size_t)n * NUM_ACT + (k - X_DIM)];
  xg[i] = (_Float16)v;
}

// Wt[n][kp] = (kp < K) ? W[kp][n] : 0   (W is [K,N] row-major fp32)
__global__ void k_transpose_w(const float* __restrict__ W,
                              _Float16* __restrict__ Wt,
                              int K, int Nout, int Kpad) {
  int i = blockIdx.x * blockDim.x + threadIdx.x;
  if (i >= Nout * Kpad) return;
  int n = i / Kpad, kp = i % Kpad;
  Wt[i] = (kp < K) ? (_Float16)W[(size_t)kp * Nout + n] : (_Float16)0.0f;
}

// ---------------------------------------------------------------------------
// WMMA GEMM: Y[M,Nout] = act(X[M,K] @ Wt^T + b), Wt stored [Nout,K] row-major.
// Block = 4 waves (128 thr), 2x2 wave grid, each wave a 32x32 tile -> 64x64.
// ---------------------------------------------------------------------------
__global__ void __launch_bounds__(128)
k_gemm_bias_act(const _Float16* __restrict__ X, const _Float16* __restrict__ Wt,
                const float* __restrict__ bias, _Float16* __restrict__ Y,
                int K, int Nout, int relu) {
  const int wave = threadIdx.x >> 5;
  const int lane = threadIdx.x & 31;
  const int rowbase = blockIdx.x * 64 + (wave >> 1) * 32;
  const int colbase = blockIdx.y * 64 + (wave & 1) * 32;

  const _Float16* Xa0 = X + (size_t)rowbase * K;
  const _Float16* Xa1 = X + (size_t)(rowbase + 16) * K;
  const _Float16* Wb0 = Wt + (size_t)colbase * K;
  const _Float16* Wb1 = Wt + (size_t)(colbase + 16) * K;

  float8 c00 = {}, c01 = {}, c10 = {}, c11 = {};

  for (int k = 0; k < K; k += 32) {
    __builtin_prefetch(Xa0 + k + 64, 0, 1);   // global_prefetch_b8
    __builtin_prefetch(Wb0 + k + 64, 0, 1);
    half16 a0 = load_a_frag(Xa0 + k, K, lane);
    half16 a1 = load_a_frag(Xa1 + k, K, lane);
    half16 b0 = load_b_frag(Wb0 + k, K, lane);
    half16 b1 = load_b_frag(Wb1 + k, K, lane);
    c00 = wmma_f16(a0, b0, c00);
    c01 = wmma_f16(a0, b1, c01);
    c10 = wmma_f16(a1, b0, c10);
    c11 = wmma_f16(a1, b1, c11);
  }

  // Epilogue. C layout: col = lane&15; rows = (lane>=16 ? 8 : 0) + vgpr index.
  const int lcol  = lane & 15;
  const int lrow0 = (lane >> 4) * 8;
  float8 accs[2][2] = {{c00, c01}, {c10, c11}};
#pragma unroll
  for (int fr = 0; fr < 2; ++fr)
#pragma unroll
    for (int fc = 0; fc < 2; ++fc) {
      int col   = colbase + fc * 16 + lcol;
      float bv  = bias[col];
      float8 ac = accs[fr][fc];
#pragma unroll
      for (int v = 0; v < 8; ++v) {
        float val = ac[v] + bv;
        if (relu) val = fmaxf(val, 0.0f);
        int row = rowbase + fr * 16 + lrow0 + v;
        Y[(size_t)row * Nout + col] = (_Float16)val;
      }
    }
}

// ---------------------------------------------------------------------------
// Row squared norms of zpos (f16 [N, Z_DIM]) -> f32
// ---------------------------------------------------------------------------
__global__ void k_row_sqnorm(const _Float16* __restrict__ Z,
                             float* __restrict__ sq) {
  int r = blockIdx.x * blockDim.x + threadIdx.x;
  if (r >= NN) return;
  const _Float16* p = Z + (size_t)r * Z_DIM;
  float acc = 0.0f;
  for (int d = 0; d < Z_DIM; ++d) {
    float v = (float)p[d];
    acc += v * v;
  }
  sq[r] = acc;
}

// ---------------------------------------------------------------------------
// Fused logits GEMM + online log-softmax, column-split for occupancy.
//   l[i][j] = (2 * z_i . zpos_j - ||zpos_j||^2) / T   (row const cancels)
// Grid = (NN/64 row blocks, NSEG col segments); one wave = 16 rows x 1024
// cols. A fragments (K=128) stay resident in VGPRs. zpos tiles are staged
// into per-wave LDS double buffers with GLOBAL_LOAD_ASYNC_TO_LDS_B128 and
// consumed via ds_load_b128; transpose each 16x16 WMMA result through LDS so
// 16 lanes own per-row running (max, sum). Partials go to workspace.
// ---------------------------------------------------------------------------
__global__ void __launch_bounds__(128)
k_loss_partial(const _Float16* __restrict__ z, const _Float16* __restrict__ zpos,
               const float* __restrict__ sqp, float* __restrict__ pm,
               float* __restrict__ ps, float* __restrict__ dg) {
  const int wave = threadIdx.x >> 5;
  const int lane = threadIdx.x & 31;
  const int rowbase = blockIdx.x * 64 + wave * 16;
  const int seg     = blockIdx.y;
  const int colbase = seg * SEG_COLS;

  __shared__ _Float16 lsb[4][2][16 * Z_DIM];  // per-wave double buffer (32 KB)
  __shared__ float    ls[4][16][17];          // per-wave transpose buffer

  // Resident A fragments: 16 rows x K=128 of z
  half16 afr[4];
#pragma unroll
  for (int kk = 0; kk < 4; ++kk)
    afr[kk] = load_a_frag(z + (size_t)rowbase * Z_DIM + kk * 32, Z_DIM, lane);

  const int lcol  = lane & 15;
  const int lrow0 = (lane >> 4) * 8;
  const int kb    = (lane >> 4) << 4;         // B-fragment K sub-offset

  unsigned lb0 = (unsigned)(size_t)&lsb[wave][0][0];  // LDS offset (addr[31:0])
  unsigned lb1 = (unsigned)(size_t)&lsb[wave][1][0];

  float m = -INFINITY, ssum = 0.0f, diag = 0.0f;

  // Prologue: stage first tile
  async_tile_load(zpos + (size_t)colbase * Z_DIM, lb0, lane);

  int cur = 0;
  for (int jt = 0; jt < SEG_CHUNK; ++jt) {
    const int jb = colbase + jt * 16;
    if (jt + 1 < SEG_CHUNK) {
      // stage next tile into the other buffer, then wait for current tile
      async_tile_load(zpos + (size_t)(jb + 16) * Z_DIM, cur ? lb0 : lb1, lane);
      wait_async_le8();
    } else {
      wait_async_0();
    }

    const _Float16* tb = &lsb[wave][cur][0];
    float8 acc = {};
#pragma unroll
    for (int kk = 0; kk < 4; ++kk) {
      const _Float16* p = tb + lcol * Z_DIM + kk * 32 + kb;
      half16 b = frag_cat(*reinterpret_cast<const half8*>(p),
                          *reinterpret_cast<const half8*>(p + 8));
      acc = wmma_f16(afr[kk], b, acc);
    }

    const float sp = sqp[jb + lcol];
#pragma unroll
    for (int v = 0; v < 8; ++v)
      ls[wave][lrow0 + v][lcol] = (2.0f * acc[v] - sp) * INV_T;
    __syncthreads();

    if (lane < 16) {
      const int rowg = rowbase + lane;
#pragma unroll
      for (int c = 0; c < 16; ++c) {
        float val = ls[wave][lane][c];
        float nm  = fmaxf(m, val);
        ssum = ssum * __expf(m - nm) + __expf(val - nm);
        m = nm;
        if (jb + c == rowg) diag = val;
      }
    }
    __syncthreads();
    cur ^= 1;
  }

  if (lane < 16) {
    const int rowg = rowbase + lane;
    pm[seg * NN + rowg] = m;
    ps[seg * NN + rowg] = ssum;
    if ((rowg / SEG_COLS) == seg) dg[rowg] = diag;   // diagonal owner
  }
}

// Merge NSEG partial (max, sum) per row, reduce loss to a scalar. Single block.
__global__ void __launch_bounds__(256)
k_finalize(const float* __restrict__ pm, const float* __restrict__ ps,
           const float* __restrict__ dg, float* __restrict__ out) {
  __shared__ float red[256];
  float acc = 0.0f;
  for (int r = threadIdx.x; r < NN; r += 256) {
    float M = pm[r];
#pragma unroll
    for (int s = 1; s < NSEG; ++s) M = fmaxf(M, pm[s * NN + r]);
    float S = 0.0f;
#pragma unroll
    for (int s = 0; s < NSEG; ++s) S += ps[s * NN + r] * __expf(pm[s * NN + r] - M);
    acc += (M + __logf(S) - dg[r]);
  }
  red[threadIdx.x] = acc;
  __syncthreads();
  for (int st = 128; st > 0; st >>= 1) {
    if (threadIdx.x < st) red[threadIdx.x] += red[threadIdx.x + st];
    __syncthreads();
  }
  if (threadIdx.x == 0) out[0] = red[0] * (1.0f / (float)NN);
}

// ---------------------------------------------------------------------------
// Host-side launcher
// ---------------------------------------------------------------------------
extern "C" void kernel_launch(void* const* d_in, const int* in_sizes, int n_in,
                              void* d_out, int out_size, void* d_ws, size_t ws_size,
                              hipStream_t stream) {
  (void)in_sizes; (void)n_in; (void)out_size; (void)ws_size;
  const float* s   = (const float*)d_in[0];
  const float* sp  = (const float*)d_in[1];
  const float* a1h = (const float*)d_in[2];
  const float* pW0 = (const float*)d_in[3];  const float* pb0 = (const float*)d_in[4];
  const float* pW1 = (const float*)d_in[5];  const float* pb1 = (const float*)d_in[6];
  const float* pW2 = (const float*)d_in[7];  const float* pb2 = (const float*)d_in[8];
  const float* gW0 = (const float*)d_in[9];  const float* gb0 = (const float*)d_in[10];
  const float* gW1 = (const float*)d_in[11]; const float* gb1 = (const float*)d_in[12];
  const float* gW2 = (const float*)d_in[13]; const float* gb2 = (const float*)d_in[14];
  float* out = (float*)d_out;

  // Workspace carve-up (256B aligned)
  char* w = (char*)d_ws;
  size_t off = 0;
  auto carve = [&](size_t bytes) {
    void* p = w + off;
    off = (off + bytes + 255) & ~(size_t)255;
    return p;
  };
  _Float16* s_h   = (_Float16*)carve((size_t)NN * X_DIM * 2);
  _Float16* xg    = (_Float16*)carve((size_t)NN * KG_PAD * 2);
  _Float16* h1    = (_Float16*)carve((size_t)NN * HDIM * 2);
  _Float16* h2    = (_Float16*)carve((size_t)NN * HDIM * 2);
  _Float16* z_h   = (_Float16*)carve((size_t)NN * Z_DIM * 2);
  _Float16* zp_h  = (_Float16*)carve((size_t)NN * Z_DIM * 2);
  float*    sqp   = (float*)   carve((size_t)NN * 4);
  float*    pm    = (float*)   carve((size_t)NSEG * NN * 4);
  float*    ps    = (float*)   carve((size_t)NSEG * NN * 4);
  float*    dg    = (float*)   carve((size_t)NN * 4);
  _Float16* pW0t  = (_Float16*)carve((size_t)HDIM * X_DIM * 2);
  _Float16* pW1t  = (_Float16*)carve((size_t)HDIM * HDIM * 2);
  _Float16* pW2t  = (_Float16*)carve((size_t)Z_DIM * HDIM * 2);
  _Float16* gW0t  = (_Float16*)carve((size_t)HDIM * KG_PAD * 2);
  _Float16* gW1t  = (_Float16*)carve((size_t)HDIM * HDIM * 2);
  _Float16* gW2t  = (_Float16*)carve((size_t)Z_DIM * HDIM * 2);

  const int TPB = 256;
  auto blocks = [](int n, int t) { return (n + t - 1) / t; };

  // --- prep -----------------------------------------------------------------
  k_cvt_f16<<<blocks(NN * X_DIM, TPB), TPB, 0, stream>>>(s, s_h, NN * X_DIM);
  k_build_xg<<<blocks(NN * KG_PAD, TPB), TPB, 0, stream>>>(sp, a1h, xg, NN * KG_PAD);
  k_transpose_w<<<blocks(HDIM * X_DIM,  TPB), TPB, 0, stream>>>(pW0, pW0t, X_DIM, HDIM, X_DIM);
  k_transpose_w<<<blocks(HDIM * HDIM,   TPB), TPB, 0, stream>>>(pW1, pW1t, HDIM, HDIM, HDIM);
  k_transpose_w<<<blocks(Z_DIM * HDIM,  TPB), TPB, 0, stream>>>(pW2, pW2t, HDIM, Z_DIM, HDIM);
  k_transpose_w<<<blocks(HDIM * KG_PAD, TPB), TPB, 0, stream>>>(gW0, gW0t, X_DIM + NUM_ACT, HDIM, KG_PAD);
  k_transpose_w<<<blocks(HDIM * HDIM,   TPB), TPB, 0, stream>>>(gW1, gW1t, HDIM, HDIM, HDIM);
  k_transpose_w<<<blocks(Z_DIM * HDIM,  TPB), TPB, 0, stream>>>(gW2, gW2t, HDIM, Z_DIM, HDIM);

  // --- MLP phi: s -> z ------------------------------------------------------
  dim3 blk(128);
  k_gemm_bias_act<<<dim3(NN / 64, HDIM / 64),  blk, 0, stream>>>(s_h, pW0t, pb0, h1, X_DIM, HDIM, 1);
  k_gemm_bias_act<<<dim3(NN / 64, HDIM / 64),  blk, 0, stream>>>(h1,  pW1t, pb1, h2, HDIM,  HDIM, 1);
  k_gemm_bias_act<<<dim3(NN / 64, Z_DIM / 64), blk, 0, stream>>>(h2,  pW2t, pb2, z_h, HDIM, Z_DIM, 0);

  // --- MLP g: [sp, a1h] -> zpos --------------------------------------------
  k_gemm_bias_act<<<dim3(NN / 64, HDIM / 64),  blk, 0, stream>>>(xg, gW0t, gb0, h1, KG_PAD, HDIM, 1);
  k_gemm_bias_act<<<dim3(NN / 64, HDIM / 64),  blk, 0, stream>>>(h1, gW1t, gb1, h2, HDIM,   HDIM, 1);
  k_gemm_bias_act<<<dim3(NN / 64, Z_DIM / 64), blk, 0, stream>>>(h2, gW2t, gb2, zp_h, HDIM, Z_DIM, 0);

  // --- fused logits GEMM + softmax loss ------------------------------------
  k_row_sqnorm<<<blocks(NN, TPB), TPB, 0, stream>>>(zp_h, sqp);
  k_loss_partial<<<dim3(NN / 64, NSEG), blk, 0, stream>>>(z_h, zp_h, sqp, pm, ps, dg);
  k_finalize<<<1, 256, 0, stream>>>(pm, ps, dg, out);
}
